// DualChannelAMPClassifier_29008209118007
// MI455X (gfx1250) — compile-verified, matrix operand order checked by reading, not run on hardware
//
#include <hip/hip_runtime.h>
#include <math.h>

// ---------------------------------------------------------------------------
// DualChannelAMPClassifier for MI455X (gfx1250, wave32, WMMA).
// All dense math runs through V_WMMA_F32_16X16X32_F16 (f16 in, f32 accum).
// GEMM tiles are staged with GLOBAL_LOAD_ASYNC_TO_LDS_B128 (ASYNCcnt) when
// fully in-range, falling back to guarded scalar staging on tails.
// ---------------------------------------------------------------------------

typedef _Float16 f16;
typedef _Float16 v16h __attribute__((ext_vector_type(16)));
typedef float    v8f  __attribute__((ext_vector_type(8)));

#define B_SZ   128
#define L_RES  128
#define L_ATM  512
#define N_RES  (B_SZ * L_RES)     // 16384
#define N_ATM  (B_SZ * L_ATM)     // 65536
#define RESDIM 1280
#define ATMDIM 21
#define HD     128
#define NHEADS 4
#define EDG    524288

// ----------------------- async global -> LDS helpers -----------------------
// ISA §10.2: for a generic pointer into LDS, addr[31:0] IS the LDS byte
// address, so the low dword of a __shared__ pointer feeds VDST directly.
__device__ __forceinline__ void async_ld_b128(void* lds_ptr, const void* gptr) {
  unsigned loff = (unsigned)(unsigned long long)lds_ptr;
  unsigned long long ga = (unsigned long long)gptr;
  asm volatile("global_load_async_to_lds_b128 %0, %1, off"
               :
               : "v"(loff), "v"(ga)
               : "memory");
}
__device__ __forceinline__ void wait_async0() {
  asm volatile("s_wait_asynccnt 0x0" ::: "memory");
}

// ------------------------- WMMA fragment helpers ---------------------------
// ISA 16-bit A-matrix 16x32 layout (CDNA5 ISA): lane l holds row M=l%16,
// VGPR v holds K = {2v, 2v+1} (+8 for lanes 16..31), v>=4 adds +8.
__device__ __forceinline__ int kmap(int g, int v) {
  return (v < 4) ? (2 * v + 8 * g) : (2 * v + 8 + 8 * g);
}

__device__ __forceinline__ v16h load_frag_A(const f16* base, int ld) {
  int lane = threadIdx.x & 31;
  int m = lane & 15, g = lane >> 4;
  const f16* row = base + m * ld;
  v16h f;
#pragma unroll
  for (int v = 0; v < 8; ++v) {
    int k = kmap(g, v);
    f[2 * v]     = row[k];
    f[2 * v + 1] = row[k + 1];
  }
  return f;
}

// B stored K-major in LDS: tile[k][n], base points at [k=0][n=n0]
__device__ __forceinline__ v16h load_frag_B_kmajor(const f16* base, int ld) {
  int lane = threadIdx.x & 31;
  int n = lane & 15, g = lane >> 4;
  v16h f;
#pragma unroll
  for (int v = 0; v < 8; ++v) {
    int k = kmap(g, v);
    f[2 * v]     = base[k * ld + n];
    f[2 * v + 1] = base[(k + 1) * ld + n];
  }
  return f;
}

// B stored N-major (rows are the N dimension, contiguous in K): tile[n][k]
__device__ __forceinline__ v16h load_frag_B_nmajor(const f16* base, int ld) {
  int lane = threadIdx.x & 31;
  int n = lane & 15, g = lane >> 4;
  const f16* row = base + n * ld;
  v16h f;
#pragma unroll
  for (int v = 0; v < 8; ++v) {
    int k = kmap(g, v);
    f[2 * v]     = row[k];
    f[2 * v + 1] = row[k + 1];
  }
  return f;
}

__device__ __forceinline__ v8f wmma16(v16h a, v16h b, v8f c) {
  return __builtin_amdgcn_wmma_f32_16x16x32_f16(false, a, false, b, (short)0, c,
                                                false, false);
}

// ------------------------- generic tiled WMMA GEMM -------------------------
// C[M,N] = act(Asrc[M,K] @ Bw[K,N] + bias + resid16)
// AMODE 0: A plain f16 [M,K]
// AMODE 1: A = concat(A [M,K1], A2 [M,K-K1])
// AMODE 2: A = concat(X[ei[m]][0:din], X[ej[m]][0:din], dist[m])  (K = 2*din+1)
// ACT 0: none, 1: relu, 2: elu
template <int AMODE, int ACT>
__global__ __launch_bounds__(256) void gemm_wmma_kernel(
    const f16* __restrict__ A, const f16* __restrict__ A2,
    const int* __restrict__ gei, const int* __restrict__ gej,
    const f16* __restrict__ dist, int din, int K1,
    const f16* __restrict__ Bw, const float* __restrict__ bias,
    const f16* __restrict__ resid16, float* __restrict__ C32,
    f16* __restrict__ C16, int M, int N, int K) {
  __shared__ f16 As[64 * 40];   // 64 x 32, padded (row stride 80B = 5*16B)
  __shared__ f16 Bs[32 * 72];   // 32 x 64, padded (row stride 144B = 9*16B)

  const int tid  = threadIdx.x;
  const int wave = tid >> 5;
  const int lane = tid & 31;
  const int n0   = blockIdx.x * 64;
  const int m0   = blockIdx.y * 64;
  const int mr   = wave & 3;          // 16-row block within tile
  const int nc   = (wave >> 2) * 32;  // 32-col block within tile

  v8f acc0 = {0.f, 0.f, 0.f, 0.f, 0.f, 0.f, 0.f, 0.f};
  v8f acc1 = {0.f, 0.f, 0.f, 0.f, 0.f, 0.f, 0.f, 0.f};

  const int arow = tid >> 2;            // 0..63
  const int ac0  = (tid & 3) * 8;       // 0,8,16,24  (16B chunks)
  const long am  = (long)(m0 + arow);
  int nodei = 0, nodej = 0;
  if (AMODE == 2) { nodei = gei[am]; nodej = gej[am]; }

  const int bk = tid >> 3;              // 0..31
  const int bn = (tid & 7) * 8;         // 0..56      (16B chunks)

  // async fast-path legality (uniform across block)
  const bool aAlign =
      (AMODE == 0) ? (K % 8 == 0)
                   : (AMODE == 1 ? (K1 % 8 == 0 && (K - K1) % 8 == 0) : false);
  const bool bAlign = (N % 8 == 0);

  for (int kt = 0; kt < K; kt += 32) {
    const bool aFast = aAlign && (kt + 32 <= K);
    const bool bFast = bAlign && (kt + 32 <= K) && (n0 + 64 <= N);

    // ---- stage A (64x32) ----
    if (aFast) {
      int k = kt + ac0;
      const f16* src;
      if (AMODE == 1 && k >= K1)
        src = A2 + am * (long)(K - K1) + (k - K1);
      else
        src = A + am * (long)(AMODE == 1 ? K1 : K) + k;
      async_ld_b128(&As[arow * 40 + ac0], src);
    } else {
#pragma unroll
      for (int j = 0; j < 8; ++j) {
        int k = kt + ac0 + j;
        f16 v = (f16)0.f;
        if (AMODE == 0) {
          if (k < K) v = A[am * K + k];
        } else if (AMODE == 1) {
          if (k < K1) v = A[am * K1 + k];
          else if (k < K) v = A2[am * (long)(K - K1) + (k - K1)];
        } else {
          if (k < din) v = A[(long)nodei * din + k];
          else if (k < 2 * din) v = A[(long)nodej * din + (k - din)];
          else if (k == 2 * din) v = dist[am];
        }
        As[arow * 40 + ac0 + j] = v;
      }
    }

    // ---- stage B (32x64) ----
    if (bFast) {
      async_ld_b128(&Bs[bk * 72 + bn], &Bw[(long)(kt + bk) * N + n0 + bn]);
    } else {
      if (kt + 32 < K)  // prefetch next K-slice of B (global_prefetch_b8)
        __builtin_prefetch(&Bw[(long)(kt + 32 + bk) * N + n0 + bn], 0, 1);
#pragma unroll
      for (int j = 0; j < 8; ++j) {
        int n = bn + j;
        f16 v = (f16)0.f;
        if (kt + bk < K && n0 + n < N) v = Bw[(long)(kt + bk) * N + (n0 + n)];
        Bs[bk * 72 + n] = v;
      }
    }
    if (aFast || bFast) wait_async0();  // s_wait_asynccnt 0
    __syncthreads();

    v16h a  = load_frag_A(&As[(mr * 16) * 40], 40);
    v16h b0 = load_frag_B_kmajor(&Bs[nc], 72);
    v16h b1 = load_frag_B_kmajor(&Bs[nc + 16], 72);
    acc0 = wmma16(a, b0, acc0);
    acc1 = wmma16(a, b1, acc1);
    __syncthreads();
  }

  // ---- epilogue ----
  const int g  = lane >> 4;
  const int nl = lane & 15;
#pragma unroll
  for (int r = 0; r < 8; ++r) {
    int m = m0 + mr * 16 + r + 8 * g;
    if (m >= M) continue;
#pragma unroll
    for (int t = 0; t < 2; ++t) {
      int n = n0 + nc + t * 16 + nl;
      if (n >= N) continue;
      float v = t ? acc1[r] : acc0[r];
      if (bias) v += bias[n];
      if (resid16) v += (float)resid16[(long)m * N + n];
      if (ACT == 1) v = fmaxf(v, 0.f);
      else if (ACT == 2) v = (v > 0.f) ? v : (__expf(v) - 1.f);
      if (C32) C32[(long)m * N + n] = v;
      if (C16) C16[(long)m * N + n] = (f16)v;
    }
  }
}

// ---------------- fused multi-head attention (L=128, d=32) -----------------
// One block per (batch, head). Scores computed with one K=32 WMMA step,
// softmax kept register-resident (shfl_xor over the 16-lane row group),
// P staged to LDS as f16, then P@V with WMMA.
__global__ __launch_bounds__(256) void mha128_kernel(
    const f16* __restrict__ Q, const f16* __restrict__ Kx,
    const f16* __restrict__ V, float* __restrict__ O) {
  const int L = 128, D = 32;
  int bh = blockIdx.x;
  int b = bh / NHEADS, h = bh % NHEADS;

  __shared__ f16 Qs[128 * 36];
  __shared__ f16 Ks[128 * 36];
  __shared__ f16 Vs[128 * 36];
  __shared__ f16 Ps[8][16 * 132];

  int tid = threadIdx.x;
  for (int i = tid; i < L * D; i += 256) {
    int r = i >> 5, c = i & 31;
    long src = ((long)(b * L + r)) * HD + h * D + c;
    Qs[r * 36 + c] = Q[src];
    Ks[r * 36 + c] = Kx[src];
    Vs[r * 36 + c] = V[src];
  }
  __syncthreads();

  int wave = tid >> 5, lane = tid & 31;
  int g = lane >> 4, nl = lane & 15;

  v16h aq = load_frag_A(&Qs[wave * 16 * 36], 36);
  v8f S[8];
#pragma unroll
  for (int j = 0; j < 8; ++j) {
    v16h bk = load_frag_B_nmajor(&Ks[j * 16 * 36], 36);
    v8f z = {0.f, 0.f, 0.f, 0.f, 0.f, 0.f, 0.f, 0.f};
    S[j] = wmma16(aq, bk, z);
  }
  const float scale = 0.17677669529663687f;  // 1/sqrt(32)
#pragma unroll
  for (int j = 0; j < 8; ++j)
#pragma unroll
    for (int r = 0; r < 8; ++r) S[j][r] *= scale;

  // row softmax: row r lives in reg r across the 16 lanes of this half-wave
#pragma unroll
  for (int r = 0; r < 8; ++r) {
    float mx = -1e30f;
#pragma unroll
    for (int j = 0; j < 8; ++j) mx = fmaxf(mx, S[j][r]);
    for (int msk = 8; msk >= 1; msk >>= 1)
      mx = fmaxf(mx, __shfl_xor(mx, msk, 32));
    float sum = 0.f;
#pragma unroll
    for (int j = 0; j < 8; ++j) {
      float e = __expf(S[j][r] - mx);
      S[j][r] = e;
      sum += e;
    }
    for (int msk = 8; msk >= 1; msk >>= 1) sum += __shfl_xor(sum, msk, 32);
    float inv = 1.f / sum;
#pragma unroll
    for (int j = 0; j < 8; ++j) S[j][r] *= inv;
  }

  // scatter P (f16) into this wave's LDS slab in row-major [16][128]
#pragma unroll
  for (int r = 0; r < 8; ++r) {
    int rl = r + 8 * g;
#pragma unroll
    for (int j = 0; j < 8; ++j)
      Ps[wave][rl * 132 + j * 16 + nl] = (f16)S[j][r];
  }
  __syncthreads();

  v8f o0 = {0.f, 0.f, 0.f, 0.f, 0.f, 0.f, 0.f, 0.f};
  v8f o1 = {0.f, 0.f, 0.f, 0.f, 0.f, 0.f, 0.f, 0.f};
#pragma unroll
  for (int ks = 0; ks < 4; ++ks) {
    v16h ap  = load_frag_A(&Ps[wave][ks * 32], 132);
    v16h bv0 = load_frag_B_kmajor(&Vs[(ks * 32) * 36 + 0], 36);
    v16h bv1 = load_frag_B_kmajor(&Vs[(ks * 32) * 36 + 16], 36);
    o0 = wmma16(ap, bv0, o0);
    o1 = wmma16(ap, bv1, o1);
  }
#pragma unroll
  for (int r = 0; r < 8; ++r) {
    int m = wave * 16 + r + 8 * g;
    O[((long)(b * L + m)) * HD + h * D + 0 + nl]  = o0[r];
    O[((long)(b * L + m)) * HD + h * D + 16 + nl] = o1[r];
  }
}

// ------------------------------- layernorm ---------------------------------
__global__ __launch_bounds__(256) void ln_kernel(
    const float* __restrict__ x, const float* __restrict__ resid,
    const float* __restrict__ gw, const float* __restrict__ bw,
    float* __restrict__ out32, f16* __restrict__ out16, int M) {
  int wave = threadIdx.x >> 5, lane = threadIdx.x & 31;
  long row = (long)blockIdx.x * 8 + wave;
  if (row >= M) return;
  const float* xr = x + row * HD;
  const float* rr = resid ? resid + row * HD : nullptr;
  float v[4], s = 0.f;
#pragma unroll
  for (int i = 0; i < 4; ++i) {
    int c = lane + 32 * i;
    v[i] = xr[c] + (rr ? rr[c] : 0.f);
    s += v[i];
  }
  for (int m = 16; m >= 1; m >>= 1) s += __shfl_xor(s, m, 32);
  float mean = s * (1.f / HD), var = 0.f;
#pragma unroll
  for (int i = 0; i < 4; ++i) {
    float d = v[i] - mean;
    var += d * d;
  }
  for (int m = 16; m >= 1; m >>= 1) var += __shfl_xor(var, m, 32);
  float inv = rsqrtf(var * (1.f / HD) + 1e-5f);
#pragma unroll
  for (int i = 0; i < 4; ++i) {
    int c = lane + 32 * i;
    float o = (v[i] - mean) * inv * gw[c] + bw[c];
    if (out32) out32[row * HD + c] = o;
    if (out16) out16[row * HD + c] = (f16)o;
  }
}

// -------------------- mean/max pooling -> concat f16 -----------------------
__global__ void pool_kernel(const float* __restrict__ x, f16* __restrict__ out,
                            int L) {
  long idx = (long)blockIdx.x * 256 + threadIdx.x;
  if (idx >= (long)B_SZ * HD) return;
  int bb = (int)(idx / HD), c = (int)(idx % HD);
  const float* p = x + ((long)bb * L) * HD + c;
  float s = 0.f, mx = -1e30f;
  for (int i = 0; i < L; ++i) {
    float v = p[(long)i * HD];
    s += v;
    mx = fmaxf(mx, v);
  }
  out[(long)bb * 2 * HD + c]      = (f16)(s / (float)L);
  out[(long)bb * 2 * HD + HD + c] = (f16)mx;
}

// ------------------------------ EGNN helpers -------------------------------
__global__ void edge_geom_kernel(const float* __restrict__ pos,
                                 const int* __restrict__ gei,
                                 const int* __restrict__ gej,
                                 float* __restrict__ pos_ij,
                                 f16* __restrict__ dist) {
  int e = blockIdx.x * 256 + threadIdx.x;
  if (e >= EDG) return;
  int i = gei[e], j = gej[e];
  float dx = pos[3 * j] - pos[3 * i];
  float dy = pos[3 * j + 1] - pos[3 * i + 1];
  float dz = pos[3 * j + 2] - pos[3 * i + 2];
  pos_ij[3 * e] = dx;
  pos_ij[3 * e + 1] = dy;
  pos_ij[3 * e + 2] = dz;
  dist[e] = (f16)sqrtf(dx * dx + dy * dy + dz * dz);
}

__global__ void edge_count_kernel(const int* __restrict__ gei,
                                  float* __restrict__ cnt) {
  int e = blockIdx.x * 256 + threadIdx.x;
  if (e >= EDG) return;
  atomicAdd(&cnt[gei[e]], 1.f);
}

__global__ void edge_agg_kernel(const float* __restrict__ msg,
                                const float* __restrict__ pos_ij,
                                const int* __restrict__ gei,
                                float* __restrict__ agg,
                                float* __restrict__ pos_upd) {
  int e = blockIdx.x * 256 + threadIdx.x;
  if (e >= EDG) return;
  const float* m = msg + (long)e * HD;
  float* a = agg + (long)gei[e] * HD;
  float nrm = 0.f;
#pragma unroll 4
  for (int c = 0; c < HD; c += 4) {
    float4 v = *(const float4*)(m + c);
    atomicAdd(&a[c], v.x);
    atomicAdd(&a[c + 1], v.y);
    atomicAdd(&a[c + 2], v.z);
    atomicAdd(&a[c + 3], v.w);
    nrm += v.x * v.x + v.y * v.y + v.z * v.z + v.w * v.w;
  }
  float w = tanhf(sqrtf(nrm));
  float* pu = pos_upd + (long)gei[e] * 3;
  atomicAdd(&pu[0], w * pos_ij[3 * e]);
  atomicAdd(&pu[1], w * pos_ij[3 * e + 1]);
  atomicAdd(&pu[2], w * pos_ij[3 * e + 2]);
}

__global__ void pos_update_kernel(const float* __restrict__ pos,
                                  const float* __restrict__ pos_upd,
                                  const float* __restrict__ cnt,
                                  float* __restrict__ pos_out) {
  int i = blockIdx.x * 256 + threadIdx.x;
  if (i >= N_ATM) return;
  float c = fmaxf(cnt[i], 1.f);
  pos_out[3 * i]     = pos[3 * i]     + pos_upd[3 * i]     / c;
  pos_out[3 * i + 1] = pos[3 * i + 1] + pos_upd[3 * i + 1] / c;
  pos_out[3 * i + 2] = pos[3 * i + 2] + pos_upd[3 * i + 2] / c;
}

__global__ void cvt16_kernel(const float* __restrict__ s, f16* __restrict__ d,
                             long n) {
  long i = (long)blockIdx.x * 256 + threadIdx.x;
  if (i < n) d[i] = (f16)s[i];
}

// ---------------------------- host-side helpers ----------------------------
template <int AMODE>
static void launch_gemm(hipStream_t st, const f16* A, const f16* A2,
                        const int* gei, const int* gej, const f16* dist,
                        int din, int K1, const f16* Bw, const float* bias,
                        const f16* resid, float* C32, f16* C16, int M, int N,
                        int K, int act) {
  dim3 g((N + 63) / 64, (M + 63) / 64);
  if (act == 0)
    gemm_wmma_kernel<AMODE, 0><<<g, 256, 0, st>>>(A, A2, gei, gej, dist, din,
                                                  K1, Bw, bias, resid, C32, C16,
                                                  M, N, K);
  else if (act == 1)
    gemm_wmma_kernel<AMODE, 1><<<g, 256, 0, st>>>(A, A2, gei, gej, dist, din,
                                                  K1, Bw, bias, resid, C32, C16,
                                                  M, N, K);
  else
    gemm_wmma_kernel<AMODE, 2><<<g, 256, 0, st>>>(A, A2, gei, gej, dist, din,
                                                  K1, Bw, bias, resid, C32, C16,
                                                  M, N, K);
}

static inline void gemm_plain(hipStream_t st, const f16* A, const f16* Bw,
                              const float* bias, const f16* resid, float* C32,
                              f16* C16, int M, int N, int K, int act) {
  launch_gemm<0>(st, A, nullptr, nullptr, nullptr, nullptr, 0, 0, Bw, bias,
                 resid, C32, C16, M, N, K, act);
}
static inline void gemm_concat(hipStream_t st, const f16* A, const f16* A2,
                               int K1, const f16* Bw, const float* bias,
                               float* C32, f16* C16, int M, int N, int K,
                               int act) {
  launch_gemm<1>(st, A, A2, nullptr, nullptr, nullptr, 0, K1, Bw, bias, nullptr,
                 C32, C16, M, N, K, act);
}
static inline void gemm_edge(hipStream_t st, const f16* X, const int* gei,
                             const int* gej, const f16* dist, int din,
                             const f16* Bw, const float* bias, f16* C16, int M,
                             int N, int K, int act) {
  launch_gemm<2>(st, X, nullptr, gei, gej, dist, din, 0, Bw, bias, nullptr,
                 nullptr, C16, M, N, K, act);
}

struct EgnnScratch {
  float* pos_ij;
  f16* dist;
  f16* msg_h;
  float* msg32;
  float* agg32;
  f16* agg16;
  f16* upd_h;
  float* pos_upd;
};

static void egnn_layer(hipStream_t st, const f16* xfeat, int din,
                       const f16* xproj, const float* pos_in, float* pos_out,
                       const f16* Wm1, const float* bm1, const f16* Wm2,
                       const float* bm2, const f16* Wu1, const float* bu1,
                       const f16* Wu2, const float* bu2, const int* gei,
                       const int* gej, const float* cnt, const EgnnScratch& s,
                       f16* x_out) {
  const int gE = EDG / 256;
  edge_geom_kernel<<<gE, 256, 0, st>>>(pos_in, gei, gej, s.pos_ij, s.dist);
  hipMemsetAsync(s.agg32, 0, (size_t)N_ATM * HD * 4, st);
  hipMemsetAsync(s.pos_upd, 0, (size_t)N_ATM * 3 * 4, st);
  // msg = W2 @ relu(W1 @ concat(x[ei], x[ej], dist))  (gather fused into A)
  gemm_edge(st, xfeat, gei, gej, s.dist, din, Wm1, bm1, s.msg_h, EDG, 2 * HD,
            2 * din + 1, 1);
  gemm_plain(st, s.msg_h, Wm2, bm2, nullptr, s.msg32, nullptr, EDG, HD, 2 * HD,
             0);
  edge_agg_kernel<<<gE, 256, 0, st>>>(s.msg32, s.pos_ij, gei, s.agg32,
                                      s.pos_upd);
  cvt16_kernel<<<(int)(((long)N_ATM * HD + 255) / 256), 256, 0, st>>>(
      s.agg32, s.agg16, (long)N_ATM * HD);
  // x_out = elu(x_proj + W_u2 @ relu(W_u1 @ concat(x_proj, agg)))
  gemm_concat(st, xproj, s.agg16, HD, Wu1, bu1, nullptr, s.upd_h, N_ATM, 2 * HD,
              2 * HD, 1);
  gemm_plain(st, s.upd_h, Wu2, bu2, xproj, nullptr, x_out, N_ATM, HD, 2 * HD,
             2);
  pos_update_kernel<<<(N_ATM + 255) / 256, 256, 0, st>>>(pos_in, s.pos_upd, cnt,
                                                         pos_out);
}

// -------------------------------- launcher ---------------------------------
extern "C" void kernel_launch(void* const* d_in, const int* in_sizes, int n_in,
                              void* d_out, int out_size, void* d_ws,
                              size_t ws_size, hipStream_t stream) {
  (void)in_sizes; (void)n_in; (void)out_size; (void)ws_size;
  // inputs in setup_inputs() insertion (DFS) order
  const float* res_x    = (const float*)d_in[0];
  const float* atom_x   = (const float*)d_in[1];
  const float* atom_pos = (const float*)d_in[2];
  const int*   gei      = (const int*)d_in[3];
  const int*   gej      = gei + EDG;
  const float* P[80];
  for (int i = 4; i < 69; ++i) P[i] = (const float*)d_in[i];

  char* wp = (char*)d_ws;
  auto ALLOC = [&](size_t bytes) -> void* {
    void* p = (void*)wp;
    wp += (bytes + 255) & ~(size_t)255;
    return p;
  };
  auto CVT = [&](const float* src, f16* dst, long n) {
    cvt16_kernel<<<(int)((n + 255) / 256), 256, 0, stream>>>(src, dst, n);
  };
  auto WGT = [&](int idx, long n) -> f16* {
    f16* d = (f16*)ALLOC((size_t)n * 2);
    CVT(P[idx], d, n);
    return d;
  };

  // ---- weights -> f16 ----
  f16* w_resproj = WGT(4, (long)RESDIM * HD);
  f16* w_raproj  = WGT(6, HD * HD);
  f16* w_q = WGT(8, HD * HD);
  f16* w_k = WGT(10, HD * HD);
  f16* w_v = WGT(12, HD * HD);
  f16* w_o = WGT(14, HD * HD);
  f16* w_ffn1 = WGT(16, HD * 2 * HD);
  f16* w_ffn2 = WGT(18, 2 * HD * HD);
  f16* w1_m1 = WGT(24, (long)(2 * ATMDIM + 1) * 2 * HD);
  f16* w1_m2 = WGT(26, 2 * HD * HD);
  f16* w1_u1 = WGT(28, 2 * HD * 2 * HD);
  f16* w1_u2 = WGT(30, 2 * HD * HD);
  f16* w1_pr = WGT(32, ATMDIM * HD);
  f16* w2_m1 = WGT(33, (long)(2 * HD + 1) * 2 * HD);
  f16* w2_m2 = WGT(35, 2 * HD * HD);
  f16* w2_u1 = WGT(37, 2 * HD * 2 * HD);
  f16* w2_u2 = WGT(39, 2 * HD * HD);
  f16* w3_m1 = WGT(41, (long)(2 * HD + 1) * 2 * HD);
  f16* w3_m2 = WGT(43, 2 * HD * HD);
  f16* w3_u1 = WGT(45, 2 * HD * 2 * HD);
  f16* w3_u2 = WGT(47, 2 * HD * HD);
  f16* w_aproj = WGT(49, HD * HD);
  f16* w_rgp = WGT(51, 2 * HD * HD);
  f16* w_agp = WGT(53, 2 * HD * HD);
  f16* w_qv = WGT(59, HD * HD);  // qkv_mha Wv (softmax over 1 key -> a == 1)
  f16* w_qo = WGT(61, HD * HD);  // qkv_mha Wo
  f16* w_fus = WGT(63, 2 * HD * HD);
  f16* w_c1 = WGT(65, HD * (HD / 2));
  f16* w_c2 = WGT(67, (HD / 2) * 1);

  // ---- activation buffers ----
  f16* resx16 = (f16*)ALLOC((size_t)N_RES * RESDIM * 2);
  f16* h016   = (f16*)ALLOC((size_t)N_RES * HD * 2);
  float* d0_32 = (float*)ALLOC((size_t)N_RES * HD * 4);
  f16*   d0_16 = (f16*)ALLOC((size_t)N_RES * HD * 2);
  f16* q16 = (f16*)ALLOC((size_t)N_RES * HD * 2);
  f16* k16 = (f16*)ALLOC((size_t)N_RES * HD * 2);
  f16* v16 = (f16*)ALLOC((size_t)N_RES * HD * 2);
  float* attn32 = (float*)ALLOC((size_t)N_RES * HD * 4);
  float* d1_32 = (float*)ALLOC((size_t)N_RES * HD * 4);
  f16*   d1_16 = (f16*)ALLOC((size_t)N_RES * HD * 2);
  f16* ffn1_16 = (f16*)ALLOC((size_t)N_RES * 2 * HD * 2);
  float* ffn2_32 = (float*)ALLOC((size_t)N_RES * HD * 4);
  float* d2_32 = (float*)ALLOC((size_t)N_RES * HD * 4);
  f16* rescat16 = (f16*)ALLOC((size_t)B_SZ * 2 * HD * 2);
  f16* resglob16 = (f16*)ALLOC((size_t)B_SZ * HD * 2);

  f16* atomx16 = (f16*)ALLOC((size_t)N_ATM * ATMDIM * 2);
  f16* xa = (f16*)ALLOC((size_t)N_ATM * HD * 2);
  f16* xb = (f16*)ALLOC((size_t)N_ATM * HD * 2);
  f16* xc = (f16*)ALLOC((size_t)N_ATM * HD * 2);
  float* posA = (float*)ALLOC((size_t)N_ATM * 3 * 4);
  float* posB = (float*)ALLOC((size_t)N_ATM * 3 * 4);
  float* cnt = (float*)ALLOC((size_t)N_ATM * 4);
  float* xatom32 = (float*)ALLOC((size_t)N_ATM * HD * 4);
  f16* atomcat16 = (f16*)ALLOC((size_t)B_SZ * 2 * HD * 2);
  f16* atomglob16 = (f16*)ALLOC((size_t)B_SZ * HD * 2);
  f16* vbuf16 = (f16*)ALLOC((size_t)B_SZ * HD * 2);
  f16* attnout16 = (f16*)ALLOC((size_t)B_SZ * HD * 2);
  f16* fused16 = (f16*)ALLOC((size_t)B_SZ * HD * 2);
  f16* c1_16 = (f16*)ALLOC((size_t)B_SZ * (HD / 2) * 2);

  EgnnScratch es;
  es.pos_ij = (float*)ALLOC((size_t)EDG * 3 * 4);
  es.dist = (f16*)ALLOC((size_t)EDG * 2);
  es.msg_h = (f16*)ALLOC((size_t)EDG * 2 * HD * 2);
  es.msg32 = (float*)ALLOC((size_t)EDG * HD * 4);
  es.agg32 = (float*)ALLOC((size_t)N_ATM * HD * 4);
  es.agg16 = (f16*)ALLOC((size_t)N_ATM * HD * 2);
  es.upd_h = (f16*)ALLOC((size_t)N_ATM * 2 * HD * 2);
  es.pos_upd = (float*)ALLOC((size_t)N_ATM * 3 * 4);

  // =================== residue channel ===================
  CVT(res_x, resx16, (long)N_RES * RESDIM);
  gemm_plain(stream, resx16, w_resproj, P[5], nullptr, nullptr, h016, N_RES, HD,
             RESDIM, 1);                                   // relu(res_proj)
  gemm_plain(stream, h016, w_raproj, P[7], nullptr, d0_32, d0_16, N_RES, HD, HD,
             0);                                           // ra_proj
  gemm_plain(stream, d0_16, w_q, P[9], nullptr, nullptr, q16, N_RES, HD, HD, 0);
  gemm_plain(stream, d0_16, w_k, P[11], nullptr, nullptr, k16, N_RES, HD, HD, 0);
  gemm_plain(stream, d0_16, w_v, P[13], nullptr, nullptr, v16, N_RES, HD, HD, 0);
  mha128_kernel<<<B_SZ * NHEADS, 256, 0, stream>>>(q16, k16, v16, attn32);
  // attn_out @ Wo + bo, then LN(d + .)
  {
    f16* mo16 = (f16*)ALLOC((size_t)N_RES * HD * 2);
    CVT(attn32, mo16, (long)N_RES * HD);
    float* mo32 = (float*)ALLOC((size_t)N_RES * HD * 4);
    gemm_plain(stream, mo16, w_o, P[15], nullptr, mo32, nullptr, N_RES, HD, HD,
               0);
    ln_kernel<<<N_RES / 8, 256, 0, stream>>>(mo32, d0_32, P[20], P[21], d1_32,
                                             d1_16, N_RES);
  }
  gemm_plain(stream, d1_16, w_ffn1, P[17], nullptr, nullptr, ffn1_16, N_RES,
             2 * HD, HD, 1);
  gemm_plain(stream, ffn1_16, w_ffn2, P[19], nullptr, ffn2_32, nullptr, N_RES,
             HD, 2 * HD, 0);
  ln_kernel<<<N_RES / 8, 256, 0, stream>>>(ffn2_32, d1_32, P[22], P[23], d2_32,
                                           nullptr, N_RES);
  pool_kernel<<<(B_SZ * HD + 255) / 256, 256, 0, stream>>>(d2_32, rescat16,
                                                           L_RES);
  gemm_plain(stream, rescat16, w_rgp, P[52], nullptr, nullptr, resglob16, B_SZ,
             HD, 2 * HD, 0);

  // =================== atom channel (3 EGNN layers) ===================
  CVT(atom_x, atomx16, (long)N_ATM * ATMDIM);
  hipMemsetAsync(cnt, 0, (size_t)N_ATM * 4, stream);
  edge_count_kernel<<<EDG / 256, 256, 0, stream>>>(gei, cnt);
  // layer 1: x_proj = atom_x @ proj
  gemm_plain(stream, atomx16, w1_pr, nullptr, nullptr, nullptr, xa, N_ATM, HD,
             ATMDIM, 0);
  egnn_layer(stream, atomx16, ATMDIM, xa, atom_pos, posA, w1_m1, P[25], w1_m2,
             P[27], w1_u1, P[29], w1_u2, P[31], gei, gej, cnt, es, xb);
  egnn_layer(stream, xb, HD, xb, posA, posB, w2_m1, P[34], w2_m2, P[36], w2_u1,
             P[38], w2_u2, P[40], gei, gej, cnt, es, xc);
  egnn_layer(stream, xc, HD, xc, posB, posA, w3_m1, P[42], w3_m2, P[44], w3_u1,
             P[46], w3_u2, P[48], gei, gej, cnt, es, xa);
  gemm_plain(stream, xa, w_aproj, P[50], nullptr, xatom32, nullptr, N_ATM, HD,
             HD, 2);  // elu(atom_proj)
  pool_kernel<<<(B_SZ * HD + 255) / 256, 256, 0, stream>>>(xatom32, atomcat16,
                                                           L_ATM);
  gemm_plain(stream, atomcat16, w_agp, P[54], nullptr, nullptr, atomglob16,
             B_SZ, HD, 2 * HD, 0);

  // =================== fusion + classifier ===================
  // seq-len-1 cross attention: softmax over one key == 1 -> o = Wo(Wv x + bv)+bo
  gemm_plain(stream, atomglob16, w_qv, P[60], nullptr, nullptr, vbuf16, B_SZ,
             HD, HD, 0);
  gemm_plain(stream, vbuf16, w_qo, P[62], nullptr, nullptr, attnout16, B_SZ, HD,
             HD, 0);
  gemm_concat(stream, resglob16, attnout16, HD, w_fus, P[64], nullptr, fused16,
              B_SZ, HD, 2 * HD, 1);
  gemm_plain(stream, fused16, w_c1, P[66], nullptr, nullptr, c1_16, B_SZ,
             HD / 2, HD, 1);
  gemm_plain(stream, c1_16, w_c2, P[68], nullptr, (float*)d_out, nullptr, B_SZ,
             1, HD / 2, 0);
}
